// GraphConvolution_40956808135018
// MI455X (gfx1250) — matrix operand order repaired and checked by reading, not compiled, not run on hardware
//
#include <hip/hip_runtime.h>
#include <hip/hip_bf16.h>

typedef __attribute__((ext_vector_type(16))) __bf16 v16bf;
typedef __attribute__((ext_vector_type(8)))  float  v8f;

#define NN 50000
#define NE 800000
#define NTILES (NE / 16)           // 50000
#define NPART 3125                 // edge blocks of 256
#define EPSV 1e-6f

// ---------- helpers ----------
__device__ __forceinline__ __bf16 f2bf(float f) {
    unsigned u = __builtin_bit_cast(unsigned, f);
    u += 0x7FFFu + ((u >> 16) & 1u);            // round-to-nearest-even
    unsigned short s = (unsigned short)(u >> 16);
    return __builtin_bit_cast(__bf16, s);
}

// ---------- K0: zero numerator (d_out) and denominator ----------
__global__ void k_zero(float* __restrict__ num, float* __restrict__ den) {
    int i = blockIdx.x * blockDim.x + threadIdx.x;
    if (i < NN * 64) num[i] = 0.0f;
    if (i < NN)      den[i] = 0.0f;
}

// ---------- K1: pack f_w[64,128] into 16 WMMA B fragments (bf16) ----------
// fragment fi = nt*4 + ks ; element (lane, j):  N = nt*16 + (lane&15)
// K = ks*32 + (j>>3)*16 + (lane>>4)*8 + (j&7) ;  B[k][n] = f_w[n*128 + k]
__global__ void k_build_b(const float* __restrict__ f_w, __bf16* __restrict__ bfrag) {
    int i = blockIdx.x * blockDim.x + threadIdx.x;   // 0..8191
    if (i >= 16 * 32 * 16) return;
    int j    = i & 15;
    int lane = (i >> 4) & 31;
    int fi   = i >> 9;
    int nt = fi >> 2, ks = fi & 3;
    int n = nt * 16 + (lane & 15);
    int k = ks * 32 + ((j >> 3) << 4) + ((lane >> 4) << 3) + (j & 7);
    bfrag[i] = f2bf(f_w[n * 128 + k]);
}

// ---------- K2: pre-convert x to bf16 in WMMA-A-fragment order ----------
// Per node row (64 feats), per 32-feat chunk c: group g = (w>>3) of within-chunk
// index w is stored at group position ((g&1)<<1)|(g>>1)  (i.e. order 0,16,8,24),
// so that lane-half h's 16 A-elements for a k-step are contiguous 32B.
__global__ void k_pack_x(const float* __restrict__ x, __bf16* __restrict__ xbf) {
    int i = blockIdx.x * blockDim.x + threadIdx.x;   // over NN*64
    if (i >= NN * 64) return;
    int n = i >> 6;
    int f = i & 63;
    int c = f >> 5;
    int w = f & 31;
    int g = w >> 3;
    int t = w & 7;
    int sg = ((g & 1) << 1) | (g >> 1);              // 0->0, 1->2, 2->1, 3->3
    xbf[(size_t)n * 64 + c * 32 + sg * 8 + t] = f2bf(x[i]);
}

// ---------- K3: per-node partial attention dots p,q ----------
__global__ void k_node_pq(const float* __restrict__ x, const float* __restrict__ w_w,
                          float* __restrict__ p, float* __restrict__ q) {
    int n = blockIdx.x * blockDim.x + threadIdx.x;
    if (n >= NN) return;
    const float* row = x + (size_t)n * 64;
    float sp = 0.f, sq = 0.f;
#pragma unroll 8
    for (int k = 0; k < 64; ++k) {
        float v = row[k];
        sp = fmaf(v, w_w[k], sp);
        sq = fmaf(v, w_w[64 + k], sq);
    }
    p[n] = sp;
    q[n] = sq;
}

// ---------- K4: per-edge score a[e] + per-block max ----------
__global__ void k_edge_a(const float* __restrict__ p, const float* __restrict__ q,
                         const int* __restrict__ src, const int* __restrict__ tgt,
                         const float* __restrict__ w_b,
                         float* __restrict__ a, float* __restrict__ part) {
    __shared__ float sm[256];
    int tid = threadIdx.x;
    int e = blockIdx.x * 256 + tid;
    float v = -3.4e38f;
    if (e < NE) {
        v = p[src[e]] + q[tgt[e]] + w_b[0];
        a[e] = v;
    }
    sm[tid] = v;
    __syncthreads();
    for (int s = 128; s > 0; s >>= 1) {
        if (tid < s) sm[tid] = fmaxf(sm[tid], sm[tid + s]);
        __syncthreads();
    }
    if (tid == 0) part[blockIdx.x] = sm[0];
}

// ---------- K5: reduce partial maxes -> amax ----------
__global__ void k_reduce_max(const float* __restrict__ part, float* __restrict__ amax) {
    __shared__ float sm[256];
    int tid = threadIdx.x;
    float v = -3.4e38f;
    for (int i = tid; i < NPART; i += 256) v = fmaxf(v, part[i]);
    sm[tid] = v;
    __syncthreads();
    for (int s = 128; s > 0; s >>= 1) {
        if (tid < s) sm[tid] = fmaxf(sm[tid], sm[tid + s]);
        __syncthreads();
    }
    if (tid == 0) *amax = sm[0];
}

// ---------- K6: a[e] <- exp(a[e]-amax) in place; accumulate denominator ----------
__global__ void k_exp_den(float* __restrict__ a, const int* __restrict__ tgt,
                          const float* __restrict__ amax_p, float* __restrict__ den) {
    int e = blockIdx.x * blockDim.x + threadIdx.x;
    if (e >= NE) return;
    float v = __expf(a[e] - *amax_p);
    a[e] = v;
    unsafeAtomicAdd(den + tgt[e], v);
}

// ---------- K7: main fused WMMA kernel ----------
// one wave32 per 16-edge tile (grid-stride); y = relu(h @ f_w.T + f_b);
// scatter y * a_exp into num[tgt]
__global__ __launch_bounds__(256) void k_gat_main(
        const __bf16* __restrict__ xbf, const int* __restrict__ src, const int* __restrict__ tgt,
        const float* __restrict__ f_b, const __bf16* __restrict__ bfrag,
        const float* __restrict__ aexp, float* __restrict__ num) {
    const int lane   = threadIdx.x & 31;
    const int half   = lane >> 4;      // 0 or 1
    const int m16    = lane & 15;      // row / N-column index
    const int waveId = blockIdx.x * (blockDim.x >> 5) + (threadIdx.x >> 5);
    const int nWaves = gridDim.x * (blockDim.x >> 5);

    // hold all 16 B fragments in registers for the whole kernel
    v16bf B[16];
#pragma unroll
    for (int fi = 0; fi < 16; ++fi)
        B[fi] = *(const v16bf*)(bfrag + (size_t)(fi * 32 + lane) * 16);

    float bias[4];
#pragma unroll
    for (int nt = 0; nt < 4; ++nt) bias[nt] = f_b[nt * 16 + m16];

    for (int tile = waveId; tile < NTILES; tile += nWaves) {
        const int ebase = tile * 16;
        const int e_m   = ebase + m16;          // this lane's A-matrix row
        const int rs    = src[e_m];
        const int rt    = tgt[e_m];

        // gather the 4 A fragments (pre-swizzled bf16: one 32B load each)
        v16bf af[4];
#pragma unroll
        for (int ks = 0; ks < 4; ++ks) {
            const __bf16* rowbf = xbf + (size_t)((ks < 2) ? rs : rt) * 64
                                      + (ks & 1) * 32 + half * 16;
            af[ks] = *(const v16bf*)rowbf;
        }

        v8f acc[4];
#pragma unroll
        for (int nt = 0; nt < 4; ++nt) {
            v8f c;
#pragma unroll
            for (int r = 0; r < 8; ++r) c[r] = bias[nt];
            acc[nt] = c;
        }

#pragma unroll
        for (int ks = 0; ks < 4; ++ks)
#pragma unroll
            for (int nt = 0; nt < 4; ++nt)
                acc[nt] = __builtin_amdgcn_wmma_f32_16x16x32_bf16(
                    false, af[ks], false, B[nt * 4 + ks], (short)0, acc[nt], false, false);

        // softmax weights for this lane's 8 accumulator rows (m = half*8 + r)
        float aw[8];
        int   tgs[8];
#pragma unroll
        for (int r = 0; r < 8; ++r) {
            int e  = ebase + half * 8 + r;
            aw[r]  = aexp[e];
            tgs[r] = tgt[e];
        }

        // scatter: num[tgt, n] += relu(y) * a_exp
#pragma unroll
        for (int nt = 0; nt < 4; ++nt) {
#pragma unroll
            for (int r = 0; r < 8; ++r) {
                float v = acc[nt][r];
                v = (v > 0.f) ? v : 0.f;
                v *= aw[r];
                unsafeAtomicAdd(num + (size_t)tgs[r] * 64 + nt * 16 + m16, v);
            }
        }
    }
}

// ---------- K8: finalize out = num / (den + eps), in place ----------
__global__ void k_finalize(float* __restrict__ out, const float* __restrict__ den) {
    int i = blockIdx.x * blockDim.x + threadIdx.x;
    if (i < NN * 64) out[i] = out[i] / (den[i >> 6] + EPSV);
}

// ---------- host launcher ----------
extern "C" void kernel_launch(void* const* d_in, const int* in_sizes, int n_in,
                              void* d_out, int out_size, void* d_ws, size_t ws_size,
                              hipStream_t stream) {
    const float* x   = (const float*)d_in[0];
    const int*   src = (const int*)d_in[1];
    const int*   tgt = (const int*)d_in[2];
    const float* f_w = (const float*)d_in[3];
    const float* f_b = (const float*)d_in[4];
    const float* w_w = (const float*)d_in[5];
    const float* w_b = (const float*)d_in[6];
    float* out = (float*)d_out;

    // workspace layout
    char*   ws    = (char*)d_ws;
    __bf16* bfrag = (__bf16*)ws;                        // 16 KB (8192 bf16)
    __bf16* xbf   = (__bf16*)(ws + 16384);              // NN*64 bf16 = 6.4 MB
    float*  wf    = (float*)(ws + 16384 + (size_t)NN * 64 * 2);
    float*  p     = wf;                                 // NN
    float*  q     = p + NN;                             // NN
    float*  a     = q + NN;                             // NE (becomes a_exp in place)
    float*  part  = a + NE;                             // NPART (pad to 3200)
    float*  amax  = part + 3200;                        // 1 (pad to 16)
    float*  den   = amax + 16;                          // NN

    k_zero<<<(NN * 64 + 255) / 256, 256, 0, stream>>>(out, den);
    k_build_b<<<32, 256, 0, stream>>>(f_w, bfrag);
    k_pack_x<<<(NN * 64 + 255) / 256, 256, 0, stream>>>(x, xbf);
    k_node_pq<<<(NN + 255) / 256, 256, 0, stream>>>(x, w_w, p, q);
    k_edge_a<<<NPART, 256, 0, stream>>>(p, q, src, tgt, w_b, a, part);
    k_reduce_max<<<1, 256, 0, stream>>>(part, amax);
    k_exp_den<<<NPART, 256, 0, stream>>>(a, tgt, amax, den);
    k_gat_main<<<640, 256, 0, stream>>>(xbf, src, tgt, f_b, bfrag, a, out);
    k_finalize<<<(NN * 64 + 255) / 256, 256, 0, stream>>>(out, den);
}
